// MambaWorldModel_45122926412157
// MI455X (gfx1250) — compile-verified, matrix operand order checked by reading, not run on hardware
//
#include <hip/hip_runtime.h>
#include <hip/hip_bf16.h>

// ---------------------------------------------------------------------------
// MambaWorldModel fused forward for gfx1250 (CDNA5, wave32, WMMA bf16)
// ---------------------------------------------------------------------------
typedef __bf16 bf16;
typedef __attribute__((ext_vector_type(16))) __bf16 v16bf;
typedef __attribute__((ext_vector_type(8)))  __bf16 v8bf;
typedef __attribute__((ext_vector_type(8)))  float  v8f;

union Frag16 { v16bf v; v8bf h[2]; };

#define B_SZ  8192
#define LAT   64
#define DM    128
#define DI    256
#define NST   16
#define NB    2
#define ROWS  32          // batch rows per workgroup
#define NTHR  256         // 8 waves of 32

// bf16 workspace layout (element offsets), all multiples of 16
#define OFF_FUS 0                       // 128 x 96   (fusion_w padded 80->96)
#define OFF_IN  12288                   // 2 x 512x128
#define OFF_XP  143360                  // 2 x 48x256 (x_proj padded 33->48)
#define OFF_OUT 167936                  // 2 x 128x256
#define OFF_NS1 233472                  // 128x128
#define OFF_NS2 249856                  // 64x128
#define OFF_RW1 258048                  // 64x128
#define OFF_DN1 266240                  // 64x128
#define WS_ELEMS 274432

// output layout (float offsets): z_next | reward | done | h_new
#define ZOUT_OFF 0
#define RW_OFF   (B_SZ * LAT)
#define DN_OFF   (RW_OFF + B_SZ)
#define H_OFF    (DN_OFF + B_SZ)

// dynamic LDS carve (bytes)
#define L_XCATB 0                        // bf16 32x96   = 6144
#define L_X     6144                     // f32  32x128  = 16384
#define L_XNB   22528                    // bf16 32x128  = 8192
#define L_XMF   30720                    // f32  32x256  = 32768
#define L_XMB   63488                    // bf16 32x256  = 16384
#define L_ZF    79872                    // f32  32x256  = 32768
#define L_XP    112640                   // f32  32x48   = 6144
#define SMEM_BYTES 118784

// ---------------------------------------------------------------------------
// One 16x16 output tile, K-loop of v_wmma_f32_16x16x32_bf16.
// A: LDS bf16, row-major (row m, k contiguous).  B: global bf16 weight,
// row-major (row = output col n, k contiguous) -> B fragment is just W rows.
// ---------------------------------------------------------------------------
template <int K>
__device__ __forceinline__ v8f wmma_tile(const bf16* __restrict__ A, int lda,
                                         const bf16* __restrict__ B, int ldb,
                                         v8f c) {
  const int lane = threadIdx.x & 31;
  const int h8   = (lane >> 4) << 3;          // K sub-offset per lane half
  const bf16* ap = A + (lane & 15) * lda + h8;
  const bf16* bp = B + (lane & 15) * ldb + h8;
#pragma unroll
  for (int k0 = 0; k0 < K; k0 += 32) {
    Frag16 a, b;
    a.h[0] = *(const v8bf*)(ap + k0);
    a.h[1] = *(const v8bf*)(ap + k0 + 16);
    b.h[0] = *(const v8bf*)(bp + k0);
    b.h[1] = *(const v8bf*)(bp + k0 + 16);
    c = __builtin_amdgcn_wmma_f32_16x16x32_bf16(false, a.v, false, b.v,
                                                (short)0, c, false, false);
  }
  return c;
}

// ---------------------------------------------------------------------------
// Prep: convert / zero-pad all weights to bf16 into d_ws (run every call).
// ---------------------------------------------------------------------------
__global__ void prep_weights(const float* __restrict__ fus,
                             const float* __restrict__ inp,
                             const float* __restrict__ xpw,
                             const float* __restrict__ outp,
                             const float* __restrict__ ns1,
                             const float* __restrict__ ns2,
                             const float* __restrict__ rw1,
                             const float* __restrict__ dn1,
                             bf16* __restrict__ ws) {
  for (int i = blockIdx.x * blockDim.x + threadIdx.x; i < WS_ELEMS;
       i += gridDim.x * blockDim.x) {
    float v;
    if (i < OFF_IN) {                         // fusion: 128 x 96 (pad k 80->96)
      int n = i / 96, k = i % 96;
      v = (k < 80) ? fus[n * 80 + k] : 0.f;
    } else if (i < OFF_XP) {                  // in_proj: contiguous copy
      v = inp[i - OFF_IN];
    } else if (i < OFF_OUT) {                 // x_proj: pad n 33->48
      int j = i - OFF_XP;
      int blk = j / 12288, r = j % 12288;
      int n = r / 256, k = r % 256;
      v = (n < 33) ? xpw[blk * 33 * 256 + n * 256 + k] : 0.f;
    } else if (i < OFF_NS1) {
      v = outp[i - OFF_OUT];
    } else if (i < OFF_NS2) {
      v = ns1[i - OFF_NS1];
    } else if (i < OFF_RW1) {
      v = ns2[i - OFF_NS2];
    } else if (i < OFF_DN1) {
      v = rw1[i - OFF_RW1];
    } else {
      v = dn1[i - OFF_DN1];
    }
    ws[i] = (bf16)v;
  }
}

// ---------------------------------------------------------------------------
// Fused forward: one workgroup = 32 batch rows, whole network.
// ---------------------------------------------------------------------------
__global__ __launch_bounds__(NTHR)
void mamba_fused(const float* __restrict__ z_t, const int* __restrict__ action,
                 const float* __restrict__ h_in, const float* __restrict__ a_embed,
                 const float* __restrict__ fusion_b,
                 const float* __restrict__ norm_g, const float* __restrict__ norm_b,
                 const float* __restrict__ log_A, const float* __restrict__ dt_bias,
                 const float* __restrict__ Dp,
                 const float* __restrict__ ns_b1, const float* __restrict__ ns_b2,
                 const float* __restrict__ rw_b1, const float* __restrict__ rw_w2,
                 const float* __restrict__ rw_b2,
                 const float* __restrict__ dn_b1, const float* __restrict__ dn_w2,
                 const float* __restrict__ dn_b2,
                 const bf16* __restrict__ ws, float* __restrict__ out) {
  extern __shared__ unsigned char smem[];
  bf16*  sXcatB = (bf16*)(smem + L_XCATB);   // 32 x 96  fused input (bf16)
  float* sX     = (float*)(smem + L_X);      // 32 x 128 residual stream
  bf16*  sXnB   = (bf16*)(smem + L_XNB);     // 32 x 128 normed x (bf16)
  float* sXmF   = (float*)(smem + L_XMF);    // 32 x 256 x_main fp32
  bf16*  sXmB   = (bf16*)(smem + L_XMB);     // 32 x 256 x_main / y (bf16)
  float* sZF    = (float*)(smem + L_ZF);     // 32 x 256 z gate fp32
  float* sXp    = (float*)(smem + L_XP);     // 32 x 48  xp (B_t|C_t|dt|pad)

  const int t    = threadIdx.x;
  const int lane = t & 31;
  const int wid  = t >> 5;
  const int r0   = blockIdx.x * ROWS;

  // ---- stage fused input [z_t | a_emb | 0pad] as bf16 --------------------
  for (int i = t; i < ROWS * 96; i += NTHR) {
    int b = i / 96, c = i % 96, gb = r0 + b;
    float v = 0.f;
    if (c < 64)      v = z_t[gb * 64 + c];
    else if (c < 80) v = a_embed[action[gb] * 16 + (c - 64)];
    sXcatB[i] = (bf16)v;
  }
  __syncthreads();

  // ---- GEMM1: x = relu(xcat @ fusion_w.T + b)  M32 N128 K96 --------------
  {
    const bf16* W = ws + OFF_FUS;
    for (int tile = wid; tile < 2 * 8; tile += 8) {
      int mt = tile >> 3, nt = tile & 7;
      v8f acc = {0.f, 0.f, 0.f, 0.f, 0.f, 0.f, 0.f, 0.f};
      acc = wmma_tile<96>(sXcatB + mt * 16 * 96, 96, W + nt * 16 * 96, 96, acc);
      int n  = nt * 16 + (lane & 15);
      int mb = mt * 16 + (lane >> 4) * 8;
      float bv = fusion_b[n];
#pragma unroll
      for (int v = 0; v < 8; ++v) {
        float val = acc[v] + bv;
        sX[(mb + v) * 128 + n] = val > 0.f ? val : 0.f;
      }
    }
  }
  __syncthreads();

  // ======================= Mamba blocks ===================================
  for (int blk = 0; blk < NB; ++blk) {
    // ---- LayerNorm over DM=128, 8 lanes per row, then bf16 stage ---------
    {
      int r = t >> 3, seg = t & 7;
      const float* xr = sX + r * 128 + seg * 16;
      float vals[16], s = 0.f, s2 = 0.f;
#pragma unroll
      for (int k = 0; k < 16; ++k) { float v = xr[k]; vals[k] = v; s += v; s2 += v * v; }
#pragma unroll
      for (int m = 1; m < 8; m <<= 1) {
        s  += __shfl_xor(s,  m, 32);
        s2 += __shfl_xor(s2, m, 32);
      }
      float mu   = s * (1.f / 128.f);
      float var  = s2 * (1.f / 128.f) - mu * mu;
      float rstd = rsqrtf(var + 1e-5f);
      const float* g  = norm_g + blk * 128 + seg * 16;
      const float* be = norm_b + blk * 128 + seg * 16;
      bf16* dst = sXnB + r * 128 + seg * 16;
#pragma unroll
      for (int k = 0; k < 16; ++k)
        dst[k] = (bf16)((vals[k] - mu) * rstd * g[k] + be[k]);
    }
    __syncthreads();

    // ---- GEMM2: xz = xn @ in_proj.T  M32 N512 K128 -----------------------
    {
      const bf16* W = ws + OFF_IN + blk * 512 * 128;
      for (int tile = wid; tile < 2 * 32; tile += 8) {
        int mt = tile >> 5, nt = tile & 31;
        v8f acc = {0.f, 0.f, 0.f, 0.f, 0.f, 0.f, 0.f, 0.f};
        acc = wmma_tile<128>(sXnB + mt * 16 * 128, 128, W + nt * 16 * 128, 128, acc);
        int n  = nt * 16 + (lane & 15);
        int mb = mt * 16 + (lane >> 4) * 8;
#pragma unroll
        for (int v = 0; v < 8; ++v) {
          int m = mb + v;
          float val = acc[v];
          if (n < 256) { sXmF[m * 256 + n] = val; sXmB[m * 256 + n] = (bf16)val; }
          else         { sZF[m * 256 + (n - 256)] = val; }
        }
      }
    }
    __syncthreads();

    // ---- GEMM3: xp = x_main @ x_proj.T  M32 N48 K256 ---------------------
    {
      const bf16* W = ws + OFF_XP + blk * 48 * 256;
      for (int tile = wid; tile < 2 * 3; tile += 8) {
        int mt = tile / 3, nt = tile % 3;
        v8f acc = {0.f, 0.f, 0.f, 0.f, 0.f, 0.f, 0.f, 0.f};
        acc = wmma_tile<256>(sXmB + mt * 16 * 256, 256, W + nt * 16 * 256, 256, acc);
        int n  = nt * 16 + (lane & 15);
        int mb = mt * 16 + (lane >> 4) * 8;
#pragma unroll
        for (int v = 0; v < 8; ++v) sXp[(mb + v) * 48 + n] = acc[v];
      }
    }
    __syncthreads();

    // ---- SSM state update (memory-bound core): thread = channel d --------
    {
      const int d = t;                         // 0..255
      float Arow[16];
      {
        const float4* la = (const float4*)(log_A + (blk * DI + d) * NST);
#pragma unroll
        for (int q = 0; q < 4; ++q) {
          float4 lv = la[q];
          Arow[q * 4 + 0] = __expf(lv.x); Arow[q * 4 + 1] = __expf(lv.y);
          Arow[q * 4 + 2] = __expf(lv.z); Arow[q * 4 + 3] = __expf(lv.w);
        }
      }
      float dtb = dt_bias[blk * DI + d];
      float Dv  = Dp[blk * DI + d];
      const float* hsrc = h_in + (size_t)blk * B_SZ * DI * NST;
      float*       hdst = out + H_OFF + (size_t)blk * B_SZ * DI * NST;
      for (int b = 0; b < ROWS; ++b) {
        int gb = r0 + b;
        float xm = sXmF[b * 256 + d];
        float dr = sXp[b * 48 + 32] + dtb;
        float delta = (dr > 20.f) ? dr : log1pf(__expf(dr));
        const float4* hp = (const float4*)(hsrc + ((size_t)gb * DI + d) * NST);
        float4*       hq = (float4*)(hdst + ((size_t)gb * DI + d) * NST);
        float y = 0.f;
#pragma unroll
        for (int q = 0; q < 4; ++q) {
          float4 hv = hp[q];
          float4 hn;
#pragma unroll
          for (int e = 0; e < 4; ++e) {
            int n = q * 4 + e;
            float Bt = sXp[b * 48 + n];
            float Ct = sXp[b * 48 + 16 + n];
            float ab = __expf(-delta * Arow[n]);
            float hv_e = (&hv.x)[e];
            float hn_e = ab * hv_e + delta * Bt * xm;
            (&hn.x)[e] = hn_e;
            y += hn_e * Ct;
          }
          hq[q] = hn;
        }
        y += Dv * xm;
        float z = sZF[b * 256 + d];
        y *= z / (1.f + __expf(-z));           // y * silu(z)
        sXmB[b * 256 + d] = (bf16)y;           // stage for out_proj GEMM
      }
    }
    __syncthreads();

    // ---- GEMM4: x = y @ out_proj.T + residual  M32 N128 K256 -------------
    {
      const bf16* W = ws + OFF_OUT + blk * 128 * 256;
      for (int tile = wid; tile < 2 * 8; tile += 8) {
        int mt = tile >> 3, nt = tile & 7;
        v8f acc = {0.f, 0.f, 0.f, 0.f, 0.f, 0.f, 0.f, 0.f};
        acc = wmma_tile<256>(sXmB + mt * 16 * 256, 256, W + nt * 16 * 256, 256, acc);
        int n  = nt * 16 + (lane & 15);
        int mb = mt * 16 + (lane >> 4) * 8;
#pragma unroll
        for (int v = 0; v < 8; ++v) {
          int m = mb + v;
          sX[m * 128 + n] = acc[v] + sX[m * 128 + n];
        }
      }
    }
    __syncthreads();
  }

  // ======================= Heads ==========================================
  for (int i = t; i < ROWS * 128; i += NTHR) sXnB[i] = (bf16)sX[i];
  __syncthreads();

  // t1 = relu(x @ ns_w1.T + ns_b1) -> bf16 (reuse sXmB as 32x128)
  bf16* sT1 = sXmB;
  {
    const bf16* W = ws + OFF_NS1;
    for (int tile = wid; tile < 16; tile += 8) {
      int mt = tile >> 3, nt = tile & 7;
      v8f acc = {0.f, 0.f, 0.f, 0.f, 0.f, 0.f, 0.f, 0.f};
      acc = wmma_tile<128>(sXnB + mt * 16 * 128, 128, W + nt * 16 * 128, 128, acc);
      int n  = nt * 16 + (lane & 15);
      int mb = mt * 16 + (lane >> 4) * 8;
      float bv = ns_b1[n];
#pragma unroll
      for (int v = 0; v < 8; ++v) {
        float val = acc[v] + bv;
        sT1[(mb + v) * 128 + n] = (bf16)(val > 0.f ? val : 0.f);
      }
    }
  }
  __syncthreads();

  // z_next = t1 @ ns_w2.T + ns_b2 -> global   (2x4 tiles)
  {
    const bf16* W = ws + OFF_NS2;
    for (int tile = wid; tile < 8; tile += 8) {
      int mt = tile >> 2, nt = tile & 3;
      v8f acc = {0.f, 0.f, 0.f, 0.f, 0.f, 0.f, 0.f, 0.f};
      acc = wmma_tile<128>(sT1 + mt * 16 * 128, 128, W + nt * 16 * 128, 128, acc);
      int n  = nt * 16 + (lane & 15);
      int mb = mt * 16 + (lane >> 4) * 8;
      float bv = ns_b2[n];
#pragma unroll
      for (int v = 0; v < 8; ++v)
        out[ZOUT_OFF + (size_t)(r0 + mb + v) * 64 + n] = acc[v] + bv;
    }
  }

  // t2 = relu(x @ rw_w1.T), t3 = relu(x @ dn_w1.T) -> fp32 (reuse sXmF 32x128)
  {
    for (int tile = wid; tile < 16; tile += 8) {
      int isdn = tile >> 3, sub = tile & 7;
      int mt = sub >> 2, nt = sub & 3;
      const bf16* W = ws + (isdn ? OFF_DN1 : OFF_RW1) + nt * 16 * 128;
      v8f acc = {0.f, 0.f, 0.f, 0.f, 0.f, 0.f, 0.f, 0.f};
      acc = wmma_tile<128>(sXnB + mt * 16 * 128, 128, W, 128, acc);
      const float* bb = isdn ? dn_b1 : rw_b1;
      int n  = nt * 16 + (lane & 15);
      int mb = mt * 16 + (lane >> 4) * 8;
      float bv = bb[n];
#pragma unroll
      for (int v = 0; v < 8; ++v) {
        float val = acc[v] + bv;
        sXmF[(mb + v) * 128 + isdn * 64 + n] = val > 0.f ? val : 0.f;
      }
    }
  }
  __syncthreads();

  // final 64-wide dots for reward / done
  if (t < 64) {
    int r = t & 31, isdn = t >> 5;
    const float* w2  = isdn ? dn_w2 : rw_w2;
    const float* row = sXmF + r * 128 + isdn * 64;
    float acc = 0.f;
#pragma unroll
    for (int k = 0; k < 64; ++k) acc += row[k] * w2[k];
    acc += isdn ? dn_b2[0] : rw_b2[0];
    out[(isdn ? DN_OFF : RW_OFF) + r0 + r] = acc;
  }
}

// ---------------------------------------------------------------------------
extern "C" void kernel_launch(void* const* d_in, const int* in_sizes, int n_in,
                              void* d_out, int out_size, void* d_ws, size_t ws_size,
                              hipStream_t stream) {
  const float* z_t        = (const float*)d_in[0];
  const int*   action     = (const int*)d_in[1];
  const float* h          = (const float*)d_in[2];
  const float* a_embed    = (const float*)d_in[3];
  const float* fusion_w   = (const float*)d_in[4];
  const float* fusion_b   = (const float*)d_in[5];
  const float* norm_g     = (const float*)d_in[6];
  const float* norm_b     = (const float*)d_in[7];
  const float* in_proj_w  = (const float*)d_in[8];
  const float* x_proj_w   = (const float*)d_in[9];
  const float* log_A      = (const float*)d_in[10];
  const float* dt_bias    = (const float*)d_in[11];
  const float* Dp         = (const float*)d_in[12];
  const float* out_proj_w = (const float*)d_in[13];
  const float* ns_w1      = (const float*)d_in[14];
  const float* ns_b1      = (const float*)d_in[15];
  const float* ns_w2      = (const float*)d_in[16];
  const float* ns_b2      = (const float*)d_in[17];
  const float* rw_w1      = (const float*)d_in[18];
  const float* rw_b1      = (const float*)d_in[19];
  const float* rw_w2      = (const float*)d_in[20];
  const float* rw_b2      = (const float*)d_in[21];
  const float* dn_w1      = (const float*)d_in[22];
  const float* dn_b1      = (const float*)d_in[23];
  const float* dn_w2      = (const float*)d_in[24];
  const float* dn_b2      = (const float*)d_in[25];
  bf16*  ws  = (bf16*)d_ws;
  float* out = (float*)d_out;

  prep_weights<<<256, 256, 0, stream>>>(fusion_w, in_proj_w, x_proj_w, out_proj_w,
                                        ns_w1, ns_w2, rw_w1, dn_w1, ws);
  mamba_fused<<<B_SZ / ROWS, NTHR, SMEM_BYTES, stream>>>(
      z_t, action, h, a_embed, fusion_b, norm_g, norm_b, log_A, dt_bias, Dp,
      ns_b1, ns_b2, rw_b1, rw_w2, rw_b2, dn_b1, dn_w2, dn_b2, ws, out);
}
